// ScaledDotProductAttention_20736102105251
// MI455X (gfx1250) — matrix operand order repaired and checked by reading, not compiled, not run on hardware
//
#include <hip/hip_runtime.h>

// ---------------------------------------------------------------------------
// Causal + padding-masked attention for MI455X (gfx1250, wave32, WMMA).
// B=8, S=4096, D=64, fp32 in/out. Flash-attention-2 style online softmax.
// Both GEMMs use v_wmma_f32_16x16x32_f16 (f16 in, f32 accumulate).
// Row reductions use DPP16 butterflies; masking is applied post-exp (exactly
// equivalent in f32 since exp(s-10000-m) underflows to 0), and is skipped
// entirely on interior, fully-valid steps (wave-uniform fast path).
// ---------------------------------------------------------------------------

typedef __attribute__((ext_vector_type(16))) _Float16 v16h;
typedef __attribute__((ext_vector_type(8)))  _Float16 v8h;
typedef __attribute__((ext_vector_type(8)))  float    v8f;
typedef __attribute__((ext_vector_type(4)))  float    v4f;

#define B_   8
#define S_   4096
#define D_   64
#define TQ   16          // q rows per wave
#define TK   32          // keys per step
#define WAVES 4
#define KSTR (D_ + 8)    // 72 f16 = 144B row stride (16B aligned)
#define VSTR (TK + 8)    // 40 f16 = 80B  row stride (16B aligned)

static __device__ __forceinline__ v16h cat8(v8h lo, v8h hi) {
  return __builtin_shufflevector(lo, hi, 0,1,2,3,4,5,6,7,8,9,10,11,12,13,14,15);
}

// ---- 16-lane butterfly reductions ----------------------------------------
#if __has_builtin(__builtin_amdgcn_update_dpp)
#define RED_DPP 1
// DPP controls: quad_perm[1,0,3,2]=0xB1 (xor1), quad_perm[2,3,0,1]=0x4E (xor2),
// row_half_mirror=0x141 (swap 4-groups), row_mirror=0x140 (swap 8-groups).
template <int CTRL>
static __device__ __forceinline__ float dppmov(float v) {
  int u = __builtin_bit_cast(int, v);
  int r = __builtin_amdgcn_update_dpp(u, u, CTRL, 0xF, 0xF, true);
  return __builtin_bit_cast(float, r);
}
#elif __has_builtin(__builtin_amdgcn_permlane16)
#define RED_PL16 1
static __device__ __forceinline__ float plx(float v, unsigned s1, unsigned s2) {
  unsigned u = __builtin_bit_cast(unsigned, v);
  unsigned r = __builtin_amdgcn_permlane16(u, u, s1, s2, false, false);
  return __builtin_bit_cast(float, r);
}
#endif

static __device__ __forceinline__ float redmax16(float v) {
#if defined(RED_DPP)
  v = fmaxf(v, dppmov<0xB1>(v));
  v = fmaxf(v, dppmov<0x4E>(v));
  v = fmaxf(v, dppmov<0x141>(v));
  v = fmaxf(v, dppmov<0x140>(v));
#elif defined(RED_PL16)
  v = fmaxf(v, plx(v, 0x67452301u, 0xEFCDAB89u));
  v = fmaxf(v, plx(v, 0x54761032u, 0xDCFE98BAu));
  v = fmaxf(v, plx(v, 0x32107654u, 0xBA98FEDCu));
  v = fmaxf(v, plx(v, 0xFEDCBA98u, 0x76543210u));
#else
  #pragma unroll
  for (int xm = 1; xm < 16; xm <<= 1) v = fmaxf(v, __shfl_xor(v, xm, 32));
#endif
  return v;
}

static __device__ __forceinline__ float redsum16(float v) {
#if defined(RED_DPP)
  v += dppmov<0xB1>(v);
  v += dppmov<0x4E>(v);
  v += dppmov<0x141>(v);
  v += dppmov<0x140>(v);
#elif defined(RED_PL16)
  v += plx(v, 0x67452301u, 0xEFCDAB89u);
  v += plx(v, 0x54761032u, 0xDCFE98BAu);
  v += plx(v, 0x32107654u, 0xBA98FEDCu);
  v += plx(v, 0xFEDCBA98u, 0x76543210u);
#else
  #pragma unroll
  for (int xm = 1; xm < 16; xm <<= 1) v += __shfl_xor(v, xm, 32);
#endif
  return v;
}

__global__ __launch_bounds__(128)
void fa_causal_f16wmma_kernel(const float* __restrict__ Q,
                              const float* __restrict__ K,
                              const float* __restrict__ V,
                              const unsigned char* __restrict__ maskp,
                              float* __restrict__ O)
{
  // LDS staging: K (row-major f16), V transposed (d-major f16), P per wave
  __shared__ alignas(16) _Float16 sK [TK][KSTR];         // 4608 B
  __shared__ alignas(16) _Float16 sVt[D_][VSTR];         // 5120 B
  __shared__ alignas(16) _Float16 sP [WAVES][TQ][VSTR];  // 5120 B

  const int nqt    = S_ / (TQ * WAVES);        // 64 q-tiles per batch
  const int b      = blockIdx.x / nqt;
  const int qt     = blockIdx.x % nqt;
  const int q_base = qt * (TQ * WAVES);

  const int tid  = threadIdx.x;
  const int wave = tid >> 5;
  const int lane = tid & 31;
  const int h    = lane >> 4;     // lane half (A/B operand K-interleave)
  const int n    = lane & 15;     // row (A) / column (B,C) index within 16

  const int q0 = q_base + wave * TQ;           // this wave's first q row

  // --- Load Q tile once, scaled by 1/sqrt(D)=0.125, into A-operand layout.
  // A 16x32 f16 layout: lane half h, elems i -> K=16h+i (contiguous)
  v16h qa[2];
  {
    const float* qrow = Q + ((size_t)b * S_ + (size_t)(q0 + n)) * D_;
    #pragma unroll
    for (int c = 0; c < 2; ++c) {
      const int base = 32 * c + 16 * h;
      #pragma unroll
      for (int i = 0; i < 8; ++i) {
        qa[c][i]     = (_Float16)(qrow[base + i]      * 0.125f);
        qa[c][i + 8] = (_Float16)(qrow[base + 8 + i] * 0.125f);
      }
    }
  }

  v8f acc[4] = {};                 // O accumulators, C layout (4 d-tiles)
  float mrow[8], lrow[8];          // per-row running max / sum (row = j + 8h)
  #pragma unroll
  for (int j = 0; j < 8; ++j) { mrow[j] = -INFINITY; lrow[j] = 0.0f; }

  const int kend = q_base + TQ * WAVES;        // causal bound for this block

  for (int kb0 = 0; kb0 < kend; kb0 += TK) {
    // ---- cooperative staging: 128 threads move K,V chunk (32x64 f32 -> f16)
    {
      const int kk = tid >> 2;                 // 0..31
      const int ds = (tid & 3) * 16;           // 0,16,32,48
      const float* krow = K + ((size_t)b * S_ + (size_t)(kb0 + kk)) * D_ + ds;
      const float* vrow = V + ((size_t)b * S_ + (size_t)(kb0 + kk)) * D_ + ds;
      v4f k0 = *(const v4f*)(krow + 0),  k1 = *(const v4f*)(krow + 4);
      v4f k2 = *(const v4f*)(krow + 8),  k3 = *(const v4f*)(krow + 12);
      v4f w0 = *(const v4f*)(vrow + 0),  w1 = *(const v4f*)(vrow + 4);
      v4f w2 = *(const v4f*)(vrow + 8),  w3 = *(const v4f*)(vrow + 12);

      // K: pack to f16 vectors -> two ds_store_b128 per thread
      v8h ka, kb;
      #pragma unroll
      for (int i = 0; i < 4; ++i) {
        ka[i]     = (_Float16)k0[i];  ka[4 + i] = (_Float16)k1[i];
        kb[i]     = (_Float16)k2[i];  kb[4 + i] = (_Float16)k3[i];
      }
      *(v8h*)&sK[kk][ds]     = ka;
      *(v8h*)&sK[kk][ds + 8] = kb;

      // V: transposed scatter (scalar b16 stores; amortized over 4 waves)
      #pragma unroll
      for (int i = 0; i < 4; ++i) {
        sVt[ds + i][kk]      = (_Float16)w0[i];
        sVt[ds + 4 + i][kk]  = (_Float16)w1[i];
        sVt[ds + 8 + i][kk]  = (_Float16)w2[i];
        sVt[ds + 12 + i][kk] = (_Float16)w3[i];
      }

      // prefetch next chunk toward L2/WGP$ (global_prefetch_b8, no counters)
      if (kb0 + TK < kend) {
        __builtin_prefetch(krow + (size_t)TK * D_, 0, 1);
        __builtin_prefetch(vrow + (size_t)TK * D_, 0, 1);
      }
    }
    __syncthreads();

    if (kb0 <= q0 + TQ - 1) {      // wave-uniform: skip fully-masked steps
      // padding-mask bytes for this lane's two columns (latency hides in WMMAs)
      const unsigned char mb0 = maskp[(size_t)b * S_ + kb0 + n];
      const unsigned char mb1 = maskp[(size_t)b * S_ + kb0 + 16 + n];

      // ---- S = Q * K^T: load ALL four B fragments first so the ds_loads
      // pipeline instead of serializing load->wait->wmma.
      v16h kbf[2][2];
      #pragma unroll
      for (int t = 0; t < 2; ++t) {
        #pragma unroll
        for (int c = 0; c < 2; ++c) {
          // B 32x16 f16 layout: lane half h holds K = 32c+16h .. +15 contiguous
          const _Float16* kp = &sK[16 * t + n][32 * c + 16 * h];
          kbf[t][c] = cat8(*(const v8h*)kp, *(const v8h*)(kp + 8));
        }
      }
      v8f st[2] = {};
      #pragma unroll
      for (int c = 0; c < 2; ++c) {
        st[0] = __builtin_amdgcn_wmma_f32_16x16x32_f16(
                    false, qa[c], false, kbf[0][c], (short)0, st[0], false, false);
        st[1] = __builtin_amdgcn_wmma_f32_16x16x32_f16(
                    false, qa[c], false, kbf[1][c], (short)0, st[1], false, false);
      }

      // ---- online softmax. The running max is taken over RAW scores (any
      // m >= true max is valid); masking is applied post-exp, where it is
      // exactly a select-to-zero (exp(s-10000-m) == 0 in f32).
      const bool interior = (kb0 + TK) <= (q0 + 1);          // causal never hits
      const bool allvalid = (__all((int)(mb0 & mb1)) != 0);  // padding never hits
      const bool fastpath = interior && allvalid;            // wave-uniform
      const int  k0i = kb0 + n;
      const int  k1i = kb0 + 16 + n;

      #pragma unroll
      for (int j = 0; j < 8; ++j) {
        const float s0 = st[0][j];
        const float s1 = st[1][j];

        const float rmax  = redmax16(fmaxf(s0, s1));
        const float mnew  = fmaxf(mrow[j], rmax);
        const float alpha = __expf(mrow[j] - mnew);
        float p0 = __expf(s0 - mnew);
        float p1 = __expf(s1 - mnew);

        if (!fastpath) {
          const int qi = q0 + j + 8 * h;
          p0 = (mb0 && k0i <= qi) ? p0 : 0.0f;
          p1 = (mb1 && k1i <= qi) ? p1 : 0.0f;
        }

        const float rsum = redsum16(p0 + p1);
        lrow[j] = lrow[j] * alpha + rsum;
        mrow[j] = mnew;
        #pragma unroll
        for (int td = 0; td < 4; ++td) acc[td][j] *= alpha;

        // transpose P through per-wave LDS (C layout -> A layout)
        sP[wave][j + 8 * h][n]      = (_Float16)p0;
        sP[wave][j + 8 * h][16 + n] = (_Float16)p1;
      }

      // ---- O += P * V  (same-wave DS ops are in-order; no barrier needed)
      {
        // A 16x32 layout for P: lane row n, K = 16h..16h+15 contiguous
        const _Float16* pp = &sP[wave][n][0];
        v16h pa = cat8(*(const v8h*)(pp + 16 * h),
                       *(const v8h*)(pp + 16 * h + 8));
        // Load all four V fragments before issuing the WMMAs.
        v16h vbf[4];
        #pragma unroll
        for (int td = 0; td < 4; ++td) {
          // B 32x16 layout for V: lane col n -> d = 16td+n, K = 16h..16h+15
          const _Float16* vp = &sVt[16 * td + n][16 * h];
          vbf[td] = cat8(*(const v8h*)vp, *(const v8h*)(vp + 8));
        }
        #pragma unroll
        for (int td = 0; td < 4; ++td)
          acc[td] = __builtin_amdgcn_wmma_f32_16x16x32_f16(
                        false, pa, false, vbf[td], (short)0, acc[td], false, false);
      }
    }
    __syncthreads();   // protect sK/sVt before next step's staging
  }

  // ---- normalize and write O (row = j+8h, d = 16*td + n; coalesced per row)
  #pragma unroll
  for (int j = 0; j < 8; ++j) {
    const float inv = 1.0f / lrow[j];
    float* orow = O + ((size_t)b * S_ + (size_t)(q0 + j + 8 * h)) * D_;
    #pragma unroll
    for (int td = 0; td < 4; ++td)
      orow[16 * td + n] = acc[td][j] * inv;
  }
}

extern "C" void kernel_launch(void* const* d_in, const int* in_sizes, int n_in,
                              void* d_out, int out_size, void* d_ws, size_t ws_size,
                              hipStream_t stream) {
  (void)in_sizes; (void)n_in; (void)out_size; (void)d_ws; (void)ws_size;
  const float*         Qp = (const float*)d_in[0];
  const float*         Kp = (const float*)d_in[1];
  const float*         Vp = (const float*)d_in[2];
  const unsigned char* Mp = (const unsigned char*)d_in[3];  // jax bool = 1 byte
  float*               Op = (float*)d_out;

  dim3 grid(B_ * (S_ / (TQ * WAVES)));   // 8 * 64 = 512 workgroups
  dim3 block(128);                       // 4 waves of 32
  fa_causal_f16wmma_kernel<<<grid, block, 0, stream>>>(Qp, Kp, Vp, Mp, Op);
}
